// _MoEBlockFallback_45277545234437
// MI455X (gfx1250) — matrix operand order repaired and checked by reading, not compiled, not run on hardware
//
#include <hip/hip_runtime.h>
#include <hip/hip_bf16.h>

// ---------------------------------------------------------------------------
// MoE block for MI455X (gfx1250, wave32, WMMA).
//   Pass 1: zero out + routing counters.
//   Pass 2: gate logits (x @ gate_w + b), top-2, build per-expert token lists.
//   Pass 3: grouped GEMM per expert over gathered token rows using
//           v_wmma_f32_16x16x32_bf16, scatter-add (+bias) into out.
// Routed FLOPs = 137 GFLOP vs 550 GFLOP dense. Block tile 128x256, each of
// 8 waves owns 64x64 = 4x4 WMMA accumulators (16 WMMAs / K-step). LDS is
// double-buffered: one barrier per K-step; staging (global load + f32->bf16
// + ds_store) overlaps the previous tile's WMMAs.
// ---------------------------------------------------------------------------

#define N_TOK 8192
#define DDIM  2048
#define NEXP  8

typedef __attribute__((ext_vector_type(16))) __bf16 v16bf;
typedef __attribute__((ext_vector_type(8)))  float  v8f;

union FragU { v16bf v; unsigned int u[8]; };

__device__ __forceinline__ unsigned short f2bf(float f) {
    // round-to-nearest-even f32 -> bf16
    unsigned int u = __builtin_bit_cast(unsigned int, f);
    unsigned int r = u + 0x7FFFu + ((u >> 16) & 1u);
    return (unsigned short)(r >> 16);
}

// ------------------------- pass 1: zero init -------------------------------
__global__ void moe_zero_kernel(float* __restrict__ out, int* __restrict__ counts,
                                size_t n) {
    size_t i = (size_t)blockIdx.x * blockDim.x + threadIdx.x;
    if (i < NEXP) counts[i] = 0;
    for (; i < n; i += (size_t)gridDim.x * blockDim.x) out[i] = 0.0f;
}

// ------------------------- pass 2: gate + route ----------------------------
__global__ void moe_gate_kernel(const float* __restrict__ x,
                                const float* __restrict__ gw,
                                const float* __restrict__ gb,
                                int* __restrict__ counts,
                                int* __restrict__ lists) {
    const int wave = threadIdx.x >> 5;
    const int lane = threadIdx.x & 31;
    const int tok  = blockIdx.x * 8 + wave;            // 1024 blocks * 8 waves
    const float* xr = x + (size_t)tok * DDIM;

    float acc[NEXP];
#pragma unroll
    for (int e = 0; e < NEXP; ++e) acc[e] = 0.0f;

    for (int d = lane; d < DDIM; d += 32) {
        const float xv = xr[d];
        const float* wr = gw + (size_t)d * NEXP;
#pragma unroll
        for (int e = 0; e < NEXP; ++e) acc[e] += xv * wr[e];
    }
#pragma unroll
    for (int e = 0; e < NEXP; ++e) {
        float v = acc[e];
#pragma unroll
        for (int off = 16; off > 0; off >>= 1) v += __shfl_xor(v, off, 32);
        acc[e] = v + gb[e];
    }
    if (lane == 0) {
        int i1 = 0; float m1 = -3.0e38f;
#pragma unroll
        for (int e = 0; e < NEXP; ++e) if (acc[e] > m1) { m1 = acc[e]; i1 = e; }
        int i2 = 0; float m2 = -3.0e38f;
#pragma unroll
        for (int e = 0; e < NEXP; ++e)
            if (e != i1 && acc[e] > m2) { m2 = acc[e]; i2 = e; }
        int p1 = atomicAdd(&counts[i1], 1); lists[i1 * N_TOK + p1] = tok;
        int p2 = atomicAdd(&counts[i2], 1); lists[i2 * N_TOK + p2] = tok;
    }
}

// ------------------------- pass 3: grouped GEMM ----------------------------
__global__ void __launch_bounds__(256, 1)
moe_gemm_kernel(const float* __restrict__ x,
                const float* __restrict__ ew,
                const float* __restrict__ eb,
                const int* __restrict__ counts,
                const int* __restrict__ lists,
                float* __restrict__ out) {
    const int e   = blockIdx.z;
    const int mt  = blockIdx.y;
    const int nt  = blockIdx.x;
    const int cnt = counts[e];
    const int row0 = mt * 128;
    if (row0 >= cnt) return;
    const int valid = min(128, cnt - row0);

    __shared__ __align__(16) unsigned short As[2][128 * 34];  // [m][k0..31]
    __shared__ __align__(16) unsigned short Bs[2][256 * 34];  // [n][k0..31]
    __shared__ int toks[128];

    const int tid  = threadIdx.x;
    const int wave = tid >> 5;
    const int lane = tid & 31;
    const int wm   = (wave & 1) * 64;   // wave M origin within 128
    const int wn   = (wave >> 1) * 64;  // wave N origin within 256
    const int lmod = lane & 15;
    const int lhi  = lane >> 4;         // lane half selects K sub-range

    // staging assignments (loop-invariant)
    const int arow = tid >> 1;            // A: 128 rows, 2 threads/row
    const int aseg = (tid & 1) * 16;      //    16 floats each
    const int bk   = tid >> 3;            // B: 32 k-rows, 8 threads/row
    const int bn   = (tid & 7) * 32;      //    32 cols each

    // hoisted gather row + global base pointers
    const int myTok = (arow < valid) ? lists[e * N_TOK + row0 + arow] : -1;
    if (tid < 128) toks[tid] = (tid < valid) ? lists[e * N_TOK + row0 + tid] : -1;
    const float* xrow = x + (size_t)(myTok < 0 ? 0 : myTok) * DDIM + aseg;
    const float* wcol = ew + (size_t)e * DDIM * DDIM + nt * 256 +
                        (size_t)bk * DDIM + bn;

    const v8f vzero = {0.f, 0.f, 0.f, 0.f, 0.f, 0.f, 0.f, 0.f};
    v8f acc[4][4];
#pragma unroll
    for (int a = 0; a < 4; ++a)
#pragma unroll
        for (int b = 0; b < 4; ++b) acc[a][b] = vzero;

    // stage one 128x32 A tile + 32x256 B tile (f32 -> bf16) into buffer `buf`
    auto stage = [&](int buf, int k0) {
        // A
        float4 f0, f1, f2, f3;
        if (myTok >= 0) {
            const float4* src = (const float4*)(xrow + k0);
            f0 = src[0]; f1 = src[1]; f2 = src[2]; f3 = src[3];
        } else {
            f0 = f1 = f2 = f3 = make_float4(0.f, 0.f, 0.f, 0.f);
        }
        unsigned short* dst = &As[buf][arow * 34 + aseg];
        const float fv[16] = {f0.x, f0.y, f0.z, f0.w, f1.x, f1.y, f1.z, f1.w,
                              f2.x, f2.y, f2.z, f2.w, f3.x, f3.y, f3.z, f3.w};
#pragma unroll
        for (int j = 0; j < 16; ++j) dst[j] = f2bf(fv[j]);
        // B (store transposed: [n][k])
        const float* srcp = wcol + (size_t)k0 * DDIM;
#pragma unroll
        for (int q = 0; q < 2; ++q) {
            const float4* src = (const float4*)(srcp + q * 16);
            float4 g0 = src[0], g1 = src[1], g2 = src[2], g3 = src[3];
            const float gv[16] = {g0.x, g0.y, g0.z, g0.w, g1.x, g1.y, g1.z, g1.w,
                                  g2.x, g2.y, g2.z, g2.w, g3.x, g3.y, g3.z, g3.w};
#pragma unroll
            for (int j = 0; j < 16; ++j)
                Bs[buf][(bn + q * 16 + j) * 34 + bk] = f2bf(gv[j]);
        }
    };

    stage(0, 0);

    for (int k0 = 0; k0 < DDIM; k0 += 32) {
        const int buf = (k0 >> 5) & 1;
        __syncthreads();                      // buffer `buf` staged & visible
        if (k0 + 32 < DDIM) stage(buf ^ 1, k0 + 32);   // overlap with WMMAs

        // ---- build fragments (contiguous k = 16*lhi + s, same order for A and B) ----
        FragU afrag[4], bfrag[4];
#pragma unroll
        for (int tm = 0; tm < 4; ++tm) {
            const int m = wm + tm * 16 + lmod;
            const unsigned short* ap = &As[buf][m * 34];
#pragma unroll
            for (int j = 0; j < 8; ++j) {
                const int kk = lhi * 16 + 2 * j;
                afrag[tm].u[j] = *(const unsigned int*)(ap + kk);
            }
        }
#pragma unroll
        for (int tn = 0; tn < 4; ++tn) {
            const int n = wn + tn * 16 + lmod;
            const unsigned short* bp = &Bs[buf][n * 34 + lhi * 16];
#pragma unroll
            for (int j = 0; j < 8; ++j)
                bfrag[tn].u[j] = *(const unsigned int*)(bp + 2 * j);
        }

        // ---- 16 WMMAs per wave per K-step ----
#pragma unroll
        for (int tm = 0; tm < 4; ++tm)
#pragma unroll
            for (int tn = 0; tn < 4; ++tn)
                acc[tm][tn] = __builtin_amdgcn_wmma_f32_16x16x32_bf16(
                    false, afrag[tm].v, false, bfrag[tn].v,
                    (short)0, acc[tm][tn], false, false);
    }

    // ---- epilogue: + bias, scatter-add into out (2 experts/token) ----
#pragma unroll
    for (int tn = 0; tn < 4; ++tn) {
        const int ncol = nt * 256 + wn + tn * 16 + lmod;
        const float bias = eb[(size_t)e * DDIM + ncol];
#pragma unroll
        for (int tm = 0; tm < 4; ++tm) {
#pragma unroll
            for (int r = 0; r < 8; ++r) {
                const int m = wm + tm * 16 + lhi * 8 + r;   // C/D layout
                if (m < valid) {
                    const int trow = toks[m];
                    __hip_atomic_fetch_add(
                        &out[(size_t)trow * DDIM + ncol],
                        acc[tm][tn][r] + bias,
                        __ATOMIC_RELAXED, __HIP_MEMORY_SCOPE_AGENT);
                }
            }
        }
    }
}

// ---------------------------------------------------------------------------
extern "C" void kernel_launch(void* const* d_in, const int* in_sizes, int n_in,
                              void* d_out, int out_size, void* d_ws, size_t ws_size,
                              hipStream_t stream) {
    (void)in_sizes; (void)n_in; (void)out_size; (void)ws_size;
    const float* x  = (const float*)d_in[0];   // [N, D]
    const float* gw = (const float*)d_in[1];   // [D, E]
    const float* gb = (const float*)d_in[2];   // [E]
    const float* ew = (const float*)d_in[3];   // [E, D, D]
    const float* eb = (const float*)d_in[4];   // [E, D]
    float* out = (float*)d_out;                // [N, D]

    int* counts = (int*)d_ws;                          // 8 ints
    int* lists  = (int*)((char*)d_ws + 256);           // [E][N] ints (256 KB)

    moe_zero_kernel<<<8192, 256, 0, stream>>>(out, counts, (size_t)N_TOK * DDIM);
    moe_gate_kernel<<<N_TOK / 8, 256, 0, stream>>>(x, gw, gb, counts, lists);
    moe_gemm_kernel<<<dim3(DDIM / 256, N_TOK / 128, NEXP), 256, 0, stream>>>(
        x, ew, eb, counts, lists, out);
}
